// GCF_68513318305793
// MI455X (gfx1250) — matrix operand order repaired and checked by reading, not compiled, not run on hardware
//
#include <hip/hip_runtime.h>
#include <hip/hip_bf16.h>

#define N_USERS  200000
#define N_NODES  300000
#define EMB      100
#define NODE_F   ((size_t)N_NODES * EMB)   // 30,000,000 floats

typedef __attribute__((ext_vector_type(16))) __bf16 v16bf;
typedef __attribute__((ext_vector_type(8)))  float  v8f;

// ---------------------------------------------------------------------------
// Init: x_cur = concat(user_emb, item_emb); acc = same; x_next = 0
// ---------------------------------------------------------------------------
__global__ void k_init(const float* __restrict__ ue, const float* __restrict__ ie,
                       float* __restrict__ xcur, float* __restrict__ acc,
                       float* __restrict__ xnext) {
    size_t i = (size_t)blockIdx.x * blockDim.x + threadIdx.x;
    if (i >= NODE_F) return;
    const size_t UF = (size_t)N_USERS * EMB;
    float v = (i < UF) ? ue[i] : ie[i - UF];
    xcur[i] = v;
    acc[i]  = v;
    xnext[i] = 0.0f;
}

// ---------------------------------------------------------------------------
// COO SpMM: y[row] += val * x[col]; one thread = one nnz * 4-float chunk
// (EMB=100 -> 25 float4 chunks; 100 floats = 400B so float4 stays aligned)
// ---------------------------------------------------------------------------
__global__ void k_spmm(const int* __restrict__ row, const int* __restrict__ col,
                       const float* __restrict__ vals, const float* __restrict__ x,
                       float* __restrict__ y, int nnz) {
    unsigned t = blockIdx.x * blockDim.x + threadIdx.x;
    unsigned n = t / 25u;
    unsigned c = (t % 25u) * 4u;
    if (n >= (unsigned)nnz) return;
    float v = vals[n];
    size_t sb = (size_t)col[n] * EMB + c;
    size_t db = (size_t)row[n] * EMB + c;
    float4 xv = *(const float4*)(x + sb);
    atomicAdd(y + db + 0, v * xv.x);
    atomicAdd(y + db + 1, v * xv.y);
    atomicAdd(y + db + 2, v * xv.z);
    atomicAdd(y + db + 3, v * xv.w);
}

// acc += src; also zero `clr` (the buffer that becomes x_next in the next layer)
__global__ void k_acczero(float* __restrict__ acc, const float* __restrict__ src,
                          float* __restrict__ clr) {
    size_t i = (size_t)blockIdx.x * blockDim.x + threadIdx.x;
    if (i >= NODE_F) return;
    acc[i] += src[i];
    clr[i] = 0.0f;
}

// ---------------------------------------------------------------------------
// Pack weights (row-major [Kin, Nout] f32) into per-lane bf16 WMMA B-fragments.
// Fragment (nt,kc): 32 lanes x 16 elems contiguous. Lane l, elem e:
//   n = nt*16 + (l & 15)
//   k = kc*32 + ((l>>4)<<3) + ((e>>3)<<4) + (e&7)   (lanes 0-15: K 0..7,16..23; lanes 16-31: K 8..15,24..31)
// ---------------------------------------------------------------------------
__global__ void k_pack(const float* __restrict__ W, int Kin, int Nout,
                       int kchunks, int ntiles, __bf16* __restrict__ out) {
    int idx = blockIdx.x * blockDim.x + threadIdx.x;
    int total = ntiles * kchunks * 512;
    if (idx >= total) return;
    int frag = idx >> 9;
    int e9   = idx & 511;
    int lane = e9 >> 4;
    int e    = e9 & 15;
    int nt = frag / kchunks;
    int kc = frag % kchunks;
    int k = kc * 32 + ((lane >> 4) << 3) + ((e >> 3) << 4) + (e & 7);
    int n = nt * 16 + (lane & 15);
    float v = (k < Kin) ? W[(size_t)k * Nout + n] : 0.0f;
    out[idx] = (__bf16)v;
}

// ---------------------------------------------------------------------------
// One WMMA layer: in = this wave's [16][instride] bf16 tile in LDS,
// out = [16][outstride] (bf16 or f32). D = relu?(A*W + bias).
// A fragment per ISA 16-bit A 16x32 layout; B fragment pre-packed contiguous.
// ---------------------------------------------------------------------------
template <typename OutT>
__device__ __forceinline__ void wmma_layer(const __bf16* __restrict__ in, int instride,
                                           OutT* __restrict__ out, int outstride,
                                           const __bf16* __restrict__ packW,
                                           const float* __restrict__ bias,
                                           int kchunks, int ntiles, int lane, bool do_relu) {
    const int mr = lane & 15;
    const int hi = lane >> 4;
    for (int nt = 0; nt < ntiles; ++nt) {
        v8f c = {};
        for (int kc = 0; kc < kchunks; ++kc) {
            v16bf a;
            const __bf16* arow = in + mr * instride + kc * 32 + (hi << 3);
#pragma unroll
            for (int e = 0; e < 16; ++e) {
                int k = ((e >> 3) << 4) + (e & 7);   // V0..3 -> K0..7, V4..7 -> K16..23 (+8 if hi)
                a[e] = arow[k];
            }
            v16bf b = *(const v16bf*)(packW + (((size_t)(nt * kchunks + kc)) << 9) + (lane << 4));
            c = __builtin_amdgcn_wmma_f32_16x16x32_bf16(false, a, false, b,
                                                        (short)0, c, false, false);
        }
        int colr = (nt << 4) + mr;
        float bv = bias[colr];
#pragma unroll
        for (int v = 0; v < 8; ++v) {
            int r = v + (hi << 3);                   // C/D layout: VGPR v -> row v (+8 hi half)
            float val = c[v] + bv;
            if (do_relu) val = val > 0.0f ? val : 0.0f;
            out[r * outstride + colr] = (OutT)val;
        }
    }
}

// ---------------------------------------------------------------------------
// Fused MLP: gather final=(acc/5)[u]||[i] -> 200 (pad 224) -> 128(relu) -> 64
// -> 32 -> 1 (+biases). 4 waves/block, each wave owns 16 rows.
// ---------------------------------------------------------------------------
#define WAVES 4
__global__ __launch_bounds__(128) void k_mlp(
    const float* __restrict__ acc,
    const int* __restrict__ uIdx, const int* __restrict__ iIdx,
    const __bf16* __restrict__ pW1, const float* __restrict__ b1,
    const __bf16* __restrict__ pW4, const float* __restrict__ b4,
    const __bf16* __restrict__ pW2, const float* __restrict__ b2,
    const float* __restrict__ W3, const float* __restrict__ b3,
    const float* __restrict__ ubias, const float* __restrict__ ibias,
    float* __restrict__ out, int B) {
    __shared__ __bf16 h0[WAVES][16][224];
    __shared__ __bf16 h1[WAVES][16][128];
    __shared__ __bf16 h2[WAVES][16][64];
    __shared__ float  h3[WAVES][16][32];

    const int wave = threadIdx.x >> 5;
    const int lane = threadIdx.x & 31;
    const int rowbase = blockIdx.x * (WAVES * 16) + wave * 16;

    // Gather: h = [final[u] , final[i+N_USERS]] * (1/5), zero-padded to 224
    for (int idx = lane; idx < 16 * 224; idx += 32) {
        int r = idx / 224, f = idx % 224;
        int g = rowbase + r;
        float v = 0.0f;
        if (g < B) {
            if (f < 100)
                v = acc[(size_t)uIdx[g] * EMB + f] * 0.2f;
            else if (f < 200)
                v = acc[((size_t)iIdx[g] + N_USERS) * EMB + (f - 100)] * 0.2f;
        }
        h0[wave][r][f] = (__bf16)v;
    }

    // Layer chain (LDS ops within a wave are in-order; per-wave private tiles)
    wmma_layer<__bf16>(&h0[wave][0][0], 224, &h1[wave][0][0], 128, pW1, b1, 7, 8, lane, true);
    wmma_layer<__bf16>(&h1[wave][0][0], 128, &h2[wave][0][0],  64, pW4, b4, 4, 4, lane, false);
    wmma_layer<float >(&h2[wave][0][0],  64, &h3[wave][0][0],  32, pW2, b2, 2, 2, lane, false);

    // Final 32 -> 1 + user/item bias: lanes 0..15 each own one row
    if (lane < 16) {
        int g = rowbase + lane;
        if (g < B) {
            float s = b3[0];
#pragma unroll
            for (int k = 0; k < 32; ++k) s += h3[wave][lane][k] * W3[k];
            s += ubias[uIdx[g]] + ibias[iIdx[g] + N_USERS];
            out[g] = s;
        }
    }
}

// ---------------------------------------------------------------------------
extern "C" void kernel_launch(void* const* d_in, const int* in_sizes, int n_in,
                              void* d_out, int out_size, void* d_ws, size_t ws_size,
                              hipStream_t stream) {
    const int*   userIdx  = (const int*)d_in[0];
    const int*   itemIdx  = (const int*)d_in[1];
    const int*   adj_row  = (const int*)d_in[2];
    const int*   adj_col  = (const int*)d_in[3];
    const float* adj_vals = (const float*)d_in[4];
    const float* user_emb = (const float*)d_in[5];
    const float* item_emb = (const float*)d_in[6];
    const float* W1 = (const float*)d_in[7];   const float* b1 = (const float*)d_in[8];
    const float* W4 = (const float*)d_in[9];   const float* b4 = (const float*)d_in[10];
    const float* W2 = (const float*)d_in[11];  const float* b2 = (const float*)d_in[12];
    const float* W3 = (const float*)d_in[13];  const float* b3 = (const float*)d_in[14];
    const float* ubias = (const float*)d_in[15];
    const float* ibias = (const float*)d_in[16];

    const int B   = in_sizes[0];
    const int nnz = in_sizes[2];

    float* buf0 = (float*)d_ws;
    float* buf1 = buf0 + NODE_F;
    float* acc  = buf1 + NODE_F;
    __bf16* pW1 = (__bf16*)(acc + NODE_F);
    __bf16* pW4 = pW1 + 7 * 8 * 512;   // 200(->224) x 128
    __bf16* pW2 = pW4 + 4 * 4 * 512;   // 128 x 64

    const int TB = 256;
    // Init node features + accumulator + zeroed next-buffer
    k_init<<<(unsigned)((NODE_F + TB - 1) / TB), TB, 0, stream>>>(user_emb, item_emb,
                                                                  buf0, acc, buf1);
    // Pack weights into bf16 WMMA B-fragments
    k_pack<<<(7 * 8 * 512 + TB - 1) / TB, TB, 0, stream>>>(W1, 200, 128, 7, 8, pW1);
    k_pack<<<(4 * 4 * 512 + TB - 1) / TB, TB, 0, stream>>>(W4, 128, 64, 4, 4, pW4);
    k_pack<<<(2 * 2 * 512 + TB - 1) / TB, TB, 0, stream>>>(W2,  64, 32, 2, 2, pW2);

    // 4 LightGCN propagation layers (L2-resident: 120MB features < 192MB L2)
    float* cur = buf0;
    float* nxt = buf1;
    for (int l = 0; l < 4; ++l) {
        size_t work = (size_t)nnz * 25;
        k_spmm<<<(unsigned)((work + TB - 1) / TB), TB, 0, stream>>>(adj_row, adj_col, adj_vals,
                                                                    cur, nxt, nnz);
        k_acczero<<<(unsigned)((NODE_F + TB - 1) / TB), TB, 0, stream>>>(acc, nxt, cur);
        float* t = cur; cur = nxt; nxt = t;
    }

    // Fused gather + 4-layer MLP with bf16 WMMA
    k_mlp<<<(B + WAVES * 16 - 1) / (WAVES * 16), WAVES * 32, 0, stream>>>(
        acc, userIdx, itemIdx, pW1, b1, pW4, b4, pW2, b2, W3, b3, ubias, ibias,
        (float*)d_out, B);
}
